// NERMRCModel_37778532336367
// MI455X (gfx1250) — matrix operand order repaired and checked by reading, compiled, not run on hardware
//
#include <hip/hip_runtime.h>
#include <hip/hip_bf16.h>

typedef _Float16 half_t;
typedef __attribute__((ext_vector_type(16))) _Float16 v16h;
typedef __attribute__((ext_vector_type(8)))  _Float16 v8h;
typedef __attribute__((ext_vector_type(8)))  float    v8f;
typedef int b128_t __attribute__((vector_size(16)));   // matches builtin param

#define NB   8
#define SEQ  512
#define HID  768
#define NHD  12
#define HDD  64
#define FFD  3072
#define NLY  4
#define MTOT (NB*SEQ)   // 4096

#define AS1 __attribute__((address_space(1)))
#define AS3 __attribute__((address_space(3)))

#if __has_builtin(__builtin_amdgcn_global_load_async_to_lds_b128)
#define HAVE_ASYNC_COPY 1
#else
#define HAVE_ASYNC_COPY 0
#endif

static constexpr int LDW = 40;   // padded LDS row stride (halfs) = 80B

// 16-byte global -> LDS copy. Async (ASYNCcnt-tracked) when the toolchain
// exposes the gfx1250 builtin, otherwise load + ds_store_b128.
static __device__ inline void copy16_g2l(const half_t* g, half_t* l)
{
#if HAVE_ASYNC_COPY
    b128_t* gp = (b128_t*)g;     // reinterpret (drops const)
    b128_t* lp = (b128_t*)l;
    __builtin_amdgcn_global_load_async_to_lds_b128((AS1 b128_t*)gp, (AS3 b128_t*)lp, 0, 0);
#else
    *(v8h*)l = *(const v8h*)g;
#endif
}

// Wait until at most the newest stage (3 ops/thread) is still in flight.
static __device__ inline void async_wait_prev()
{
#if HAVE_ASYNC_COPY
    asm volatile("s_wait_asynccnt 3" ::: "memory");
#endif
}

static __device__ inline void async_wait_all()
{
#if HAVE_ASYNC_COPY
    asm volatile("s_wait_asynccnt 0" ::: "memory");
#endif
}

// Stage one 128x32 A tile and one 64x32 B tile (16B chunks, 3 per thread).
static __device__ inline void stage_tiles(const half_t* Ab, int lda, int blockM,
                                          const half_t* Bb, int ldb, int blockN,
                                          int k, half_t* Asb, half_t* Bsb, int tid)
{
    const int c0 = tid;
    copy16_g2l(Ab + (long)(blockM + (c0 >> 2)) * lda + k + ((c0 & 3) << 3),
               Asb + (c0 >> 2) * LDW + ((c0 & 3) << 3));
    const int c1 = tid + 256;
    copy16_g2l(Ab + (long)(blockM + (c1 >> 2)) * lda + k + ((c1 & 3) << 3),
               Asb + (c1 >> 2) * LDW + ((c1 & 3) << 3));
    copy16_g2l(Bb + (long)(blockN + (tid >> 2)) * ldb + k + ((tid & 3) << 3),
               Bsb + (tid >> 2) * LDW + ((tid & 3) << 3));
}

// ---------------------------------------------------------------------------
// Double-buffered, async-LDS-staged WMMA GEMM:
//   C[M,N] = A[M,K] @ Bt[N,K]^T (+bias)
// A: f16 row-major (lda), Bt: f16 transposed layout [N,K] row-major (ldb).
// Block = 256 threads = 8 waves stacked in M -> block tile 128x64.
// Each wave computes 16x64 (4 accumulators) per K-chunk of 32.
// Async DMA for chunk kc+1/kc+2 overlaps WMMA on chunk kc.
// amode/bmode/cmode==1 -> per-head batch offset (z = b*NHD + h).
// EPI: 0=f32+bias, 1=f16+bias, 2=f16 V-transpose store, 3=f16 gelu+bias
// ---------------------------------------------------------------------------
template<int EPI>
__global__ __launch_bounds__(256)
void gemm_wmma(const half_t* __restrict__ A, int lda, long sA, int amode,
               const half_t* __restrict__ Bt, int ldb, long sB, int bmode,
               const float* __restrict__ bias,
               float* __restrict__ Cf, half_t* __restrict__ Ch,
               int ldc, long sC, int cmode, int K)
{
    __shared__ half_t As[2][128 * LDW];     // 2 x (128 x 32) tiles
    __shared__ half_t Bs[2][64 * LDW];      // 2 x (64 x 32) tiles

    const int tid  = threadIdx.x;
    const int lane = tid & 31;
    const int w    = tid >> 5;              // wave id 0..7 (M direction)
    const int blockM = blockIdx.y * 128;
    const int blockN = blockIdx.x * 64;
    const int z = blockIdx.z;

    const long headOff = (long)(z / NHD) * (SEQ * (long)HID) + (long)(z % NHD) * HDD;
    const long aoff = (amode == 1) ? headOff : sA * z;
    const long boff = (bmode == 1) ? headOff : sB * z;
    const long coff = (cmode == 1) ? headOff : sC * z;

    const half_t* Ab = A  + aoff;
    const half_t* Bb = Bt + boff;

    v8f acc[4] = {};
    const int nk = K >> 5;                  // K-chunks of 32

    // ---- pipeline prologue: stage chunks 0 and 1 ----
    stage_tiles(Ab, lda, blockM, Bb, ldb, blockN, 0, As[0], Bs[0], tid);
    if (nk > 1)
        stage_tiles(Ab, lda, blockM, Bb, ldb, blockN, 32, As[1], Bs[1], tid);

    for (int kc = 0; kc < nk; ++kc) {
        const int p = kc & 1;
        if (kc + 1 < nk) async_wait_prev();   // chunk kc done (in-order), kc+1 in flight
        else             async_wait_all();
        __syncthreads();                      // cross-wave visibility of buffer p

        // ---- all fragments first, then back-to-back WMMAs ----
        const half_t* arow = &As[p][(w * 16 + (lane & 15)) * LDW + ((lane >> 4) << 3)];
        v8h a0 = *(const v8h*)(arow);
        v8h a1 = *(const v8h*)(arow + 16);
        v16h a = __builtin_shufflevector(a0, a1, 0,1,2,3,4,5,6,7,8,9,10,11,12,13,14,15);

        v16h bf[4];
        #pragma unroll
        for (int t = 0; t < 4; ++t) {
            const half_t* brow = &Bs[p][(t * 16 + (lane & 15)) * LDW + ((lane >> 4) << 3)];
            v8h b0 = *(const v8h*)(brow);
            v8h b1 = *(const v8h*)(brow + 16);
            bf[t] = __builtin_shufflevector(b0, b1, 0,1,2,3,4,5,6,7,8,9,10,11,12,13,14,15);
        }
        #pragma unroll
        for (int t = 0; t < 4; ++t)
            acc[t] = __builtin_amdgcn_wmma_f32_16x16x32_f16(false, a, false, bf[t],
                                                            (short)0, acc[t], false, false);

        __syncthreads();                      // all waves done reading buffer p
        if (kc + 2 < nk)                      // refill freed buffer (overlaps next compute)
            stage_tiles(Ab, lda, blockM, Bb, ldb, blockN, (kc + 2) * 32,
                        As[p], Bs[p], tid);
    }

    #pragma unroll
    for (int t = 0; t < 4; ++t) {
        #pragma unroll
        for (int r = 0; r < 8; ++r) {
            const int m = blockM + w * 16 + r + ((lane >> 4) << 3);
            const int n = blockN + t * 16 + (lane & 15);
            float v = acc[t][r];
            if (bias) v += bias[n];
            if (EPI == 0) {
                Cf[coff + (long)m * ldc + n] = v;
            } else if (EPI == 1) {
                Ch[coff + (long)m * ldc + n] = (half_t)v;
            } else if (EPI == 2) {
                // V-projection transposed store: out[b][h][d][l]
                // m = b*SEQ + l ; n = h*HDD + d
                Ch[(long)(m >> 9) * (HID * (long)SEQ) + (long)n * SEQ + (m & 511)] = (half_t)v;
            } else {
                float g = 0.5f * v * (1.0f + erff(v * 0.70710678118f));
                Ch[coff + (long)m * ldc + n] = (half_t)g;
            }
        }
    }
}

// ---------------------------------------------------------------------------
// f32 [K,N] -> f16 [N,K] transpose (LDS tiled, 32x32 tiles, 256 threads)
// ---------------------------------------------------------------------------
__global__ __launch_bounds__(256)
void transpose_to_f16(const float* __restrict__ in, half_t* __restrict__ out,
                      int K, int N)
{
    __shared__ float tile[32][33];
    const int n0 = blockIdx.x * 32, k0 = blockIdx.y * 32;
    const int tx = threadIdx.x & 31, ty = threadIdx.x >> 5;
    #pragma unroll
    for (int i = 0; i < 32; i += 8)
        tile[ty + i][tx] = in[(long)(k0 + ty + i) * N + n0 + tx];
    __syncthreads();
    #pragma unroll
    for (int i = 0; i < 32; i += 8)
        out[(long)(n0 + ty + i) * K + k0 + tx] = (half_t)tile[tx][ty + i];
}

// ---------------------------------------------------------------------------
// Embedding gather + LayerNorm (one 256-thread block per token row of 768)
// ---------------------------------------------------------------------------
__global__ __launch_bounds__(256)
void embed_ln(const int* __restrict__ ids, const int* __restrict__ tti,
              const float* __restrict__ we, const float* __restrict__ pe,
              const float* __restrict__ te,
              const float* __restrict__ g, const float* __restrict__ b,
              float* __restrict__ xf, half_t* __restrict__ xh)
{
    const long row = blockIdx.x;
    const int  l   = (int)(row & 511);
    const int  id  = ids[row];
    const int  tt  = tti[row];
    const int  tid = threadIdx.x;
    __shared__ float red[256];
    __shared__ float sh_m, sh_r;

    float v[3]; float s = 0.f;
    #pragma unroll
    for (int i = 0; i < 3; ++i) {
        const int c = tid + i * 256;
        float t = we[(long)id * HID + c] + pe[(long)l * HID + c] + te[(long)tt * HID + c];
        v[i] = t; s += t;
    }
    red[tid] = s; __syncthreads();
    for (int st = 128; st; st >>= 1) { if (tid < st) red[tid] += red[tid + st]; __syncthreads(); }
    if (!tid) sh_m = red[0] * (1.0f / HID);
    __syncthreads();
    const float mean = sh_m;
    float ss = 0.f;
    #pragma unroll
    for (int i = 0; i < 3; ++i) { float d = v[i] - mean; ss += d * d; }
    red[tid] = ss; __syncthreads();
    for (int st = 128; st; st >>= 1) { if (tid < st) red[tid] += red[tid + st]; __syncthreads(); }
    if (!tid) sh_r = rsqrtf(red[0] * (1.0f / HID) + 1e-12f);
    __syncthreads();
    const float rstd = sh_r;
    #pragma unroll
    for (int i = 0; i < 3; ++i) {
        const int c = tid + i * 256;
        float o = (v[i] - mean) * rstd * g[c] + b[c];
        xf[row * HID + c] = o;
        xh[row * HID + c] = (half_t)o;
    }
}

// ---------------------------------------------------------------------------
// Residual add + LayerNorm; writes f32 and f16 copies of the new hidden state
// ---------------------------------------------------------------------------
__global__ __launch_bounds__(256)
void resid_ln(const float* __restrict__ xin, const float* __restrict__ yin,
              const float* __restrict__ g, const float* __restrict__ b,
              float* __restrict__ xf, half_t* __restrict__ xh)
{
    const long row = blockIdx.x;
    const int  tid = threadIdx.x;
    __shared__ float red[256];
    __shared__ float sh_m, sh_r;

    float v[3]; float s = 0.f;
    #pragma unroll
    for (int i = 0; i < 3; ++i) {
        const int c = tid + i * 256;
        float t = xin[row * HID + c] + yin[row * HID + c];
        v[i] = t; s += t;
    }
    red[tid] = s; __syncthreads();
    for (int st = 128; st; st >>= 1) { if (tid < st) red[tid] += red[tid + st]; __syncthreads(); }
    if (!tid) sh_m = red[0] * (1.0f / HID);
    __syncthreads();
    const float mean = sh_m;
    float ss = 0.f;
    #pragma unroll
    for (int i = 0; i < 3; ++i) { float d = v[i] - mean; ss += d * d; }
    red[tid] = ss; __syncthreads();
    for (int st = 128; st; st >>= 1) { if (tid < st) red[tid] += red[tid + st]; __syncthreads(); }
    if (!tid) sh_r = rsqrtf(red[0] * (1.0f / HID) + 1e-12f);
    __syncthreads();
    const float rstd = sh_r;
    #pragma unroll
    for (int i = 0; i < 3; ++i) {
        const int c = tid + i * 256;
        float o = (v[i] - mean) * rstd * g[c] + b[c];
        xf[row * HID + c] = o;
        xh[row * HID + c] = (half_t)o;
    }
}

// ---------------------------------------------------------------------------
// Masked, scaled softmax over score rows of length 512. One wave per row.
// ---------------------------------------------------------------------------
__global__ __launch_bounds__(256)
void softmax_rows(const float* __restrict__ s, const int* __restrict__ mask,
                  half_t* __restrict__ p)
{
    const int  wid  = threadIdx.x >> 5;
    const int  lane = threadIdx.x & 31;
    const long row  = (long)blockIdx.x * 8 + wid;       // [0, NB*NHD*SEQ)
    const int  bb   = (int)(row / (NHD * SEQ));
    const float* src = s + row * SEQ;
    half_t*      dst = p + row * SEQ;
    const int*  mrow = mask + bb * SEQ;

    float vals[16];
    float mx = -1e30f;
    #pragma unroll
    for (int i = 0; i < 16; ++i) {
        const int j = lane + i * 32;
        const float mb = (1.0f - (float)mrow[j]) * -1e9f;
        const float x = src[j] * 0.125f + mb;           // 1/sqrt(64)
        vals[i] = x;
        mx = fmaxf(mx, x);
    }
    for (int o = 16; o; o >>= 1) mx = fmaxf(mx, __shfl_xor(mx, o, 32));
    float sum = 0.f;
    #pragma unroll
    for (int i = 0; i < 16; ++i) { vals[i] = expf(vals[i] - mx); sum += vals[i]; }
    for (int o = 16; o; o >>= 1) sum += __shfl_xor(sum, o, 32);
    const float inv = 1.0f / sum;
    #pragma unroll
    for (int i = 0; i < 16; ++i) dst[lane + i * 32] = (half_t)(vals[i] * inv);
}

// ---------------------------------------------------------------------------
// Span heads: 4 dot products per row (start, end, match[:H], match[H:])
// ---------------------------------------------------------------------------
__global__ __launch_bounds__(256)
void heads_kernel(const float* __restrict__ x,
                  const float* __restrict__ sw, const float* __restrict__ sb,
                  const float* __restrict__ ew, const float* __restrict__ eb,
                  const float* __restrict__ mw,
                  float* __restrict__ out,
                  float* __restrict__ sproj, float* __restrict__ eproj)
{
    const long row = blockIdx.x;
    const int  tid = threadIdx.x;
    const float* xr = x + row * HID;
    float a = 0.f, bv = 0.f, c = 0.f, d = 0.f;
    for (int i = tid; i < HID; i += 256) {
        const float xv = xr[i];
        a  += xv * sw[i];
        bv += xv * ew[i];
        c  += xv * mw[i];
        d  += xv * mw[HID + i];
    }
    __shared__ float r0[256], r1[256], r2[256], r3[256];
    r0[tid] = a; r1[tid] = bv; r2[tid] = c; r3[tid] = d;
    __syncthreads();
    for (int st = 128; st; st >>= 1) {
        if (tid < st) {
            r0[tid] += r0[tid + st]; r1[tid] += r1[tid + st];
            r2[tid] += r2[tid + st]; r3[tid] += r3[tid + st];
        }
        __syncthreads();
    }
    if (!tid) {
        out[row]        = r0[0] + sb[0];
        out[MTOT + row] = r1[0] + eb[0];
        sproj[row] = r2[0];
        eproj[row] = r3[0];
    }
}

// ---------------------------------------------------------------------------
// match_logit[b,i,j] = sproj[b,i] + eproj[b,j] + match_b
// ---------------------------------------------------------------------------
__global__ __launch_bounds__(256)
void match_kernel(const float* __restrict__ sproj, const float* __restrict__ eproj,
                  const float* __restrict__ mb, float* __restrict__ out)
{
    const long total = (long)NB * SEQ * SEQ;
    const long idx = (long)blockIdx.x * 256 + threadIdx.x;
    if (idx >= total) return;
    const int bb  = (int)(idx / (SEQ * SEQ));
    const int rem = (int)(idx % (SEQ * SEQ));
    const int i = rem >> 9, j = rem & 511;
    out[2 * MTOT + idx] = sproj[bb * SEQ + i] + eproj[bb * SEQ + j] + mb[0];
}

// ---------------------------------------------------------------------------
extern "C" void kernel_launch(void* const* d_in, const int* in_sizes, int n_in,
                              void* d_out, int out_size, void* d_ws, size_t ws_size,
                              hipStream_t stream)
{
    // ---- inputs (setup_inputs order) ----
    const int*   input_ids = (const int*)  d_in[0];
    const int*   tok_type  = (const int*)  d_in[1];
    const int*   attn_mask = (const int*)  d_in[2];
    const float* word_emb  = (const float*)d_in[3];
    const float* pos_emb   = (const float*)d_in[4];
    const float* type_emb  = (const float*)d_in[5];
    const float* emb_g     = (const float*)d_in[6];
    const float* emb_b     = (const float*)d_in[7];
    const float* Wq = (const float*)d_in[8];   const float* bq = (const float*)d_in[9];
    const float* Wk = (const float*)d_in[10];  const float* bk = (const float*)d_in[11];
    const float* Wv = (const float*)d_in[12];  const float* bv = (const float*)d_in[13];
    const float* Wo = (const float*)d_in[14];  const float* bo = (const float*)d_in[15];
    const float* attn_g = (const float*)d_in[16]; const float* attn_b = (const float*)d_in[17];
    const float* Wff1 = (const float*)d_in[18];   const float* bff1 = (const float*)d_in[19];
    const float* Wff2 = (const float*)d_in[20];   const float* bff2 = (const float*)d_in[21];
    const float* ffn_g = (const float*)d_in[22];  const float* ffn_b = (const float*)d_in[23];
    const float* start_w = (const float*)d_in[24]; const float* start_b = (const float*)d_in[25];
    const float* end_w   = (const float*)d_in[26]; const float* end_b   = (const float*)d_in[27];
    const float* match_w = (const float*)d_in[28]; const float* match_b = (const float*)d_in[29];
    float* out = (float*)d_out;

    // ---- workspace carve-up ----
    char* wsb = (char*)d_ws;
    size_t off = 0;
    auto take = [&](size_t bytes) -> void* {
        void* p = wsb + off;
        off = (off + bytes + 255) & ~(size_t)255;
        return p;
    };
    float*  xf32   = (float*) take((size_t)MTOT * HID * 4);
    half_t* xh     = (half_t*)take((size_t)MTOT * HID * 2);
    half_t* qh     = (half_t*)take((size_t)MTOT * HID * 2);
    half_t* kh     = (half_t*)take((size_t)MTOT * HID * 2);
    half_t* vth    = (half_t*)take((size_t)MTOT * HID * 2);   // [B,NH,64,512]
    half_t* ctxh   = (half_t*)take((size_t)MTOT * HID * 2);
    half_t* hh     = (half_t*)take((size_t)MTOT * FFD * 2);
    float*  scores = (float*) take((size_t)NB * NHD * SEQ * SEQ * 4);
    half_t* probs  = (half_t*)take((size_t)NB * NHD * SEQ * SEQ * 2);
    float*  yf32   = (float*) take((size_t)MTOT * HID * 4);
    float*  sproj  = (float*) take((size_t)MTOT * 4);
    float*  eproj  = (float*) take((size_t)MTOT * 4);
    half_t* wqT  = (half_t*)take((size_t)NLY * HID * HID * 2);
    half_t* wkT  = (half_t*)take((size_t)NLY * HID * HID * 2);
    half_t* wvT  = (half_t*)take((size_t)NLY * HID * HID * 2);
    half_t* woT  = (half_t*)take((size_t)NLY * HID * HID * 2);
    half_t* wf1T = (half_t*)take((size_t)NLY * HID * FFD * 2); // [3072,768] per layer
    half_t* wf2T = (half_t*)take((size_t)NLY * FFD * HID * 2); // [768,3072] per layer
    (void)ws_size; (void)n_in; (void)in_sizes; (void)out_size;

    // ---- weight conversion: f32 [K,N] -> f16 [N,K] ----
    for (int l = 0; l < NLY; ++l) {
        const long HH = (long)HID * HID, HF = (long)HID * FFD;
        transpose_to_f16<<<dim3(HID/32, HID/32), 256, 0, stream>>>(Wq + l*HH, wqT + l*HH, HID, HID);
        transpose_to_f16<<<dim3(HID/32, HID/32), 256, 0, stream>>>(Wk + l*HH, wkT + l*HH, HID, HID);
        transpose_to_f16<<<dim3(HID/32, HID/32), 256, 0, stream>>>(Wv + l*HH, wvT + l*HH, HID, HID);
        transpose_to_f16<<<dim3(HID/32, HID/32), 256, 0, stream>>>(Wo + l*HH, woT + l*HH, HID, HID);
        transpose_to_f16<<<dim3(FFD/32, HID/32), 256, 0, stream>>>(Wff1 + l*HF, wf1T + l*HF, HID, FFD);
        transpose_to_f16<<<dim3(HID/32, FFD/32), 256, 0, stream>>>(Wff2 + l*HF, wf2T + l*HF, FFD, HID);
    }

    // ---- embeddings + LN ----
    embed_ln<<<MTOT, 256, 0, stream>>>(input_ids, tok_type, word_emb, pos_emb, type_emb,
                                       emb_g, emb_b, xf32, xh);

    const long HH = (long)HID * HID, HF = (long)HID * FFD;
    for (int l = 0; l < NLY; ++l) {
        // Q, K projections (f16 out)
        gemm_wmma<1><<<dim3(HID/64, MTOT/128, 1), 256, 0, stream>>>(
            xh, HID, 0, 0, wqT + l*HH, HID, 0, 0, bq + l*HID,
            nullptr, qh, HID, 0, 0, HID);
        gemm_wmma<1><<<dim3(HID/64, MTOT/128, 1), 256, 0, stream>>>(
            xh, HID, 0, 0, wkT + l*HH, HID, 0, 0, bk + l*HID,
            nullptr, kh, HID, 0, 0, HID);
        // V projection, stored transposed per head: [B,NH,64,512]
        gemm_wmma<2><<<dim3(HID/64, MTOT/128, 1), 256, 0, stream>>>(
            xh, HID, 0, 0, wvT + l*HH, HID, 0, 0, bv + l*HID,
            nullptr, vth, HID, 0, 0, HID);
        // scores = q @ k^T  (batched over B*NH, K=64)
        gemm_wmma<0><<<dim3(SEQ/64, SEQ/128, NB*NHD), 256, 0, stream>>>(
            qh, HID, 0, 1, kh, HID, 0, 1, nullptr,
            scores, nullptr, SEQ, (long)SEQ*SEQ, 0, HDD);
        // softmax (scale + mask) -> f16 probs
        softmax_rows<<<(NB*NHD*SEQ)/8, 256, 0, stream>>>(scores, attn_mask, probs);
        // ctx = probs @ v   (batched, K=512, N=64) scattered back to [B,L,H] f16
        gemm_wmma<1><<<dim3(HDD/64, SEQ/128, NB*NHD), 256, 0, stream>>>(
            probs, SEQ, (long)SEQ*SEQ, 0, vth, SEQ, (long)HDD*SEQ, 0, nullptr,
            nullptr, ctxh, HID, 0, 1, SEQ);
        // attention output projection (f32 out)
        gemm_wmma<0><<<dim3(HID/64, MTOT/128, 1), 256, 0, stream>>>(
            ctxh, HID, 0, 0, woT + l*HH, HID, 0, 0, bo + l*HID,
            yf32, nullptr, HID, 0, 0, HID);
        resid_ln<<<MTOT, 256, 0, stream>>>(xf32, yf32, attn_g + l*HID, attn_b + l*HID, xf32, xh);
        // FFN up + exact GELU (f16 out)
        gemm_wmma<3><<<dim3(FFD/64, MTOT/128, 1), 256, 0, stream>>>(
            xh, HID, 0, 0, wf1T + l*HF, HID, 0, 0, bff1 + l*FFD,
            nullptr, hh, FFD, 0, 0, HID);
        // FFN down (f32 out)
        gemm_wmma<0><<<dim3(HID/64, MTOT/128, 1), 256, 0, stream>>>(
            hh, FFD, 0, 0, wf2T + l*HF, FFD, 0, 0, bff2 + l*HID,
            yf32, nullptr, HID, 0, 0, FFD);
        resid_ln<<<MTOT, 256, 0, stream>>>(xf32, yf32, ffn_g + l*HID, ffn_b + l*HID, xf32, xh);
    }

    // ---- heads ----
    heads_kernel<<<MTOT, 256, 0, stream>>>(xf32, start_w, start_b, end_w, end_b,
                                           match_w, out, sproj, eproj);
    const long mtotal = (long)NB * SEQ * SEQ;
    match_kernel<<<(unsigned)((mtotal + 255) / 256), 256, 0, stream>>>(sproj, eproj, match_b, out);
}
